// MultiHeadedAttention_63299228008878
// MI455X (gfx1250) — matrix operand order repaired and checked by reading, not compiled, not run on hardware
//
#include <hip/hip_runtime.h>

#define HH 16
#define DMODEL 1024
#define BB 8
#define NN 1024
#define DKK 64

typedef __attribute__((ext_vector_type(16))) _Float16 v16h;
typedef __attribute__((ext_vector_type(8)))  float    v8f;
typedef _Float16 h16;

__device__ __forceinline__ v8f wmma16(v16h a, v16h b, v8f c) {
  // D = A(16x32 f16) * B(32x16 f16) + C(16x16 f32)
  return __builtin_amdgcn_wmma_f32_16x16x32_f16(false, a, false, b, (short)0, c,
                                                false, false);
}

// A-fragment (16x32 f16), layout used here: lane group g (lane>>4) holds
// K = g*16 .. g*16+15 contiguously in elements 0..15.
__device__ __forceinline__ v16h load_a_f16(const h16* __restrict__ row, int g) {
  v16h a;
  const h16* p0 = row + g * 16;
  const h16* p1 = row + g * 16 + 8;
#pragma unroll
  for (int e = 0; e < 8; ++e) { a[e] = p0[e]; a[8 + e] = p1[e]; }
  return a;
}

__device__ __forceinline__ v16h load_a_f32cvt(const float* __restrict__ row, int g) {
  v16h a;
  const float* p0 = row + g * 16;
  const float* p1 = row + g * 16 + 8;
#pragma unroll
  for (int e = 0; e < 8; ++e) { a[e] = (h16)p0[e]; a[8 + e] = (h16)p1[e]; }
  return a;
}

// B-fragment (32x16 f16): lane holds 16 contiguous rows K=g*16..+15 of column c.
__device__ __forceinline__ v16h load_b16(const h16* __restrict__ p) {
  v16h b;
#pragma unroll
  for (int e = 0; e < 16; ++e) b[e] = p[e];
  return b;
}

// ---------------------------------------------------------------------------
// Kernel 1: QKV projections.  z=0: Q=(query@Wq+bq)*0.125 -> [b,h,n,dk] f16
//                             z=1: K=(key@Wk+bk)          -> [b,h,n,dk] f16
//                             z=2: V=(value@Wv+bv)        -> [b,h,dk,n] f16 (T)
// ---------------------------------------------------------------------------
__global__ __launch_bounds__(256) void proj_kernel(
    const float* __restrict__ Xq, const float* __restrict__ Xk,
    const float* __restrict__ Xv, const float* __restrict__ Wq,
    const float* __restrict__ bq, const float* __restrict__ Wk,
    const float* __restrict__ bk, const float* __restrict__ Wv,
    const float* __restrict__ bv, h16* __restrict__ Qo, h16* __restrict__ Ko,
    h16* __restrict__ Vt) {
  constexpr int WSTR = 40;                 // halves; 80B row stride, 16B aligned
  __shared__ h16 Wl[64 * WSTR];
  const int z = blockIdx.z;
  const float* X  = (z == 0) ? Xq : (z == 1) ? Xk : Xv;
  const float* W  = (z == 0) ? Wq : (z == 1) ? Wk : Wv;
  const float* bi = (z == 0) ? bq : (z == 1) ? bk : bv;
  const int rb = blockIdx.x, cb = blockIdx.y;
  const int wv = threadIdx.x >> 5, ln = threadIdx.x & 31;
  const int g = ln >> 4, c = ln & 15;
  const int row0 = rb * 128 + wv * 16;

  v8f acc[4];
#pragma unroll
  for (int j = 0; j < 4; ++j)
    acc[j] = (v8f){0.f, 0.f, 0.f, 0.f, 0.f, 0.f, 0.f, 0.f};

  for (int kk = 0; kk < DMODEL / 32; ++kk) {
    // stage 32(k) x 64(col) W tile into LDS, transposed + f16
#pragma unroll
    for (int i = 0; i < 8; ++i) {
      int e = threadIdx.x * 8 + i;
      int ek = e >> 6, ec = e & 63;
      Wl[ec * WSTR + ek] =
          (h16)W[(size_t)(kk * 32 + ek) * DMODEL + cb * 64 + ec];
    }
    __syncthreads();
    const float* arow = X + (size_t)(row0 + c) * DMODEL + kk * 32;
    v16h a = load_a_f32cvt(arow, g);
    // load all four B fragments first so the four WMMAs can overlap
    v16h bf0 = load_b16(&Wl[(0 * 16 + c) * WSTR + g * 16]);
    v16h bf1 = load_b16(&Wl[(1 * 16 + c) * WSTR + g * 16]);
    v16h bf2 = load_b16(&Wl[(2 * 16 + c) * WSTR + g * 16]);
    v16h bf3 = load_b16(&Wl[(3 * 16 + c) * WSTR + g * 16]);
    acc[0] = wmma16(a, bf0, acc[0]);
    acc[1] = wmma16(a, bf1, acc[1]);
    acc[2] = wmma16(a, bf2, acc[2]);
    acc[3] = wmma16(a, bf3, acc[3]);
    __syncthreads();
  }

#pragma unroll
  for (int jf = 0; jf < 4; ++jf) {
#pragma unroll
    for (int r = 0; r < 8; ++r) {
      int R = row0 + r + 8 * g;          // global row in [0, B*N)
      int bb = R >> 10, n = R & 1023;
      int d = jf * 16 + c;               // dim within head (cb == head)
      float val = acc[jf][r] + bi[cb * 64 + d];
      if (z == 0) {
        val *= 0.125f;                   // 1/sqrt(dk)
        Qo[((size_t)(bb * HH + cb) * NN + n) * DKK + d] = (h16)val;
      } else if (z == 1) {
        Ko[((size_t)(bb * HH + cb) * NN + n) * DKK + d] = (h16)val;
      } else {
        Vt[((size_t)(bb * HH + cb) * DKK + d) * NN + n] = (h16)val;
      }
    }
  }
}

// ---------------------------------------------------------------------------
// Kernel 2: fused flash attention + adjacency prior.
// One wave = 16 queries of one (b,h).  x[b,n,h*64+d] =
//   0.5 * softmax(QK^T masked) @ V  +  0.5 * (adj/rowsum) @ V
// ---------------------------------------------------------------------------
__global__ __launch_bounds__(256) void attn_kernel(
    const h16* __restrict__ Q, const h16* __restrict__ K,
    const h16* __restrict__ Vt, const float* __restrict__ adj,
    const int* __restrict__ mask, h16* __restrict__ X) {
  constexpr int PSTR = 40;
  __shared__ h16 Pb[8][16 * PSTR];
  __shared__ float RSb[8][16];
  const int b = blockIdx.z, h = blockIdx.y;
  const int wv = threadIdx.x >> 5, ln = threadIdx.x & 31;
  const int g = ln >> 4, c = ln & 15;
  const int q0 = blockIdx.x * 128 + wv * 16;

  const h16* qrow = Q + ((size_t)(b * HH + h) * NN + (q0 + c)) * DKK;
  v16h qa0 = load_a_f16(qrow, g);        // dims 0..31
  v16h qa1 = load_a_f16(qrow + 32, g);   // dims 32..63

  const v8f zf = (v8f){0.f, 0.f, 0.f, 0.f, 0.f, 0.f, 0.f, 0.f};
  v8f o[4], oa[4];
#pragma unroll
  for (int j = 0; j < 4; ++j) { o[j] = zf; oa[j] = zf; }
  float mx[8], l[8];
#pragma unroll
  for (int r = 0; r < 8; ++r) { mx[r] = -3.0e38f; l[r] = 0.f; }
  float rsacc = 0.f;
  h16* P = Pb[wv];

  for (int kt = 0; kt < NN / 32; ++kt) {
    const int k0 = kt * 32;
    const h16* kr0 = K + ((size_t)(b * HH + h) * NN + k0 + c) * DKK;
    const h16* kr1 = kr0 + 16 * DKK;
    v16h kb00 = load_b16(kr0 + g * 16);
    v16h kb01 = load_b16(kr0 + 32 + g * 16);
    v16h kb10 = load_b16(kr1 + g * 16);
    v16h kb11 = load_b16(kr1 + 32 + g * 16);
    v8f c0 = wmma16(qa1, kb01, wmma16(qa0, kb00, zf));
    v8f c1 = wmma16(qa1, kb11, wmma16(qa0, kb10, zf));

    const bool mk0 = mask[b * NN + k0 + c] != 0;        // key col = c
    const bool mk1 = mask[b * NN + k0 + 16 + c] != 0;

#pragma unroll
    for (int r = 0; r < 8; ++r) {        // row = r + 8*g (C layout)
      float s0 = mk0 ? c0[r] : -1.0e12f;
      float s1 = mk1 ? c1[r] : -1.0e12f;
      float pm = fmaxf(s0, s1);
      pm = fmaxf(pm, __shfl_xor(pm, 1));
      pm = fmaxf(pm, __shfl_xor(pm, 2));
      pm = fmaxf(pm, __shfl_xor(pm, 4));
      pm = fmaxf(pm, __shfl_xor(pm, 8));  // row max across 16-lane group
      float mo = fmaxf(mx[r], pm);
      float sc = __expf(mx[r] - mo);
      mx[r] = mo;
      float p0 = __expf(s0 - mo);
      float p1 = __expf(s1 - mo);
      float ps = p0 + p1;
      ps += __shfl_xor(ps, 1);
      ps += __shfl_xor(ps, 2);
      ps += __shfl_xor(ps, 4);
      ps += __shfl_xor(ps, 8);
      l[r] = l[r] * sc + ps;
      o[0][r] *= sc; o[1][r] *= sc; o[2][r] *= sc; o[3][r] *= sc;
      // stage P (C layout -> LDS) for A-fragment relayout
      P[(r + 8 * g) * PSTR + c]      = (h16)p0;
      P[(r + 8 * g) * PSTR + 16 + c] = (h16)p1;
    }
    v16h pa = load_a_f16(P + c * PSTR, g);   // in-wave LDS ops stay in order

    // adjacency A-fragment straight from global f32, plus running row sums
    const float* arow = adj + ((size_t)b * NN + (q0 + c)) * NN + k0;
    v16h aa;
#pragma unroll
    for (int e = 0; e < 8; ++e) {
      float f0 = arow[g * 16 + e];
      float f1 = arow[g * 16 + 8 + e];
      rsacc += f0 + f1;
      aa[e] = (h16)f0;
      aa[8 + e] = (h16)f1;
    }

    // load all four V fragments first, then issue the 8 WMMAs back-to-back
    const h16* vbase =
        Vt + ((size_t)((b * HH + h) * DKK)) * NN + k0 + g * 16 + (size_t)c * NN;
    v16h vb0 = load_b16(vbase + (size_t)(0 * 16) * NN);
    v16h vb1 = load_b16(vbase + (size_t)(1 * 16) * NN);
    v16h vb2 = load_b16(vbase + (size_t)(2 * 16) * NN);
    v16h vb3 = load_b16(vbase + (size_t)(3 * 16) * NN);
    o[0]  = wmma16(pa, vb0, o[0]);
    o[1]  = wmma16(pa, vb1, o[1]);
    o[2]  = wmma16(pa, vb2, o[2]);
    o[3]  = wmma16(pa, vb3, o[3]);
    oa[0] = wmma16(aa, vb0, oa[0]);
    oa[1] = wmma16(aa, vb1, oa[1]);
    oa[2] = wmma16(aa, vb2, oa[2]);
    oa[3] = wmma16(aa, vb3, oa[3]);
  }

  // full adj row sums: lane groups hold disjoint column halves of row c
  float rs = rsacc + __shfl_xor(rsacc, 16);
  RSb[wv][c] = rs;
  __syncthreads();

#pragma unroll
  for (int jf = 0; jf < 4; ++jf) {
#pragma unroll
    for (int r = 0; r < 8; ++r) {
      int row = r + 8 * g;
      float val = 0.5f * o[jf][r] / l[r] +
                  0.5f * oa[jf][r] / (RSb[wv][row] + 1e-6f);
      X[((size_t)b * NN + q0 + row) * DMODEL + h * DKK + jf * 16 + c] =
          (h16)val;
    }
  }
}

// ---------------------------------------------------------------------------
// Kernel 3: out = x @ Wo + bo  (f16 x, f32 W staged as f16, f32 out)
// ---------------------------------------------------------------------------
__global__ __launch_bounds__(256) void out_kernel(
    const h16* __restrict__ Xh, const float* __restrict__ Wo,
    const float* __restrict__ bo, float* __restrict__ out) {
  constexpr int WSTR = 40;
  __shared__ h16 Wl[64 * WSTR];
  const int rb = blockIdx.x, cb = blockIdx.y;
  const int wv = threadIdx.x >> 5, ln = threadIdx.x & 31;
  const int g = ln >> 4, c = ln & 15;
  const int row0 = rb * 128 + wv * 16;

  v8f acc[4];
#pragma unroll
  for (int j = 0; j < 4; ++j)
    acc[j] = (v8f){0.f, 0.f, 0.f, 0.f, 0.f, 0.f, 0.f, 0.f};

  for (int kk = 0; kk < DMODEL / 32; ++kk) {
#pragma unroll
    for (int i = 0; i < 8; ++i) {
      int e = threadIdx.x * 8 + i;
      int ek = e >> 6, ec = e & 63;
      Wl[ec * WSTR + ek] =
          (h16)Wo[(size_t)(kk * 32 + ek) * DMODEL + cb * 64 + ec];
    }
    __syncthreads();
    const h16* arow = Xh + (size_t)(row0 + c) * DMODEL + kk * 32;
    v16h a = load_a_f16(arow, g);
    v16h bf0 = load_b16(&Wl[(0 * 16 + c) * WSTR + g * 16]);
    v16h bf1 = load_b16(&Wl[(1 * 16 + c) * WSTR + g * 16]);
    v16h bf2 = load_b16(&Wl[(2 * 16 + c) * WSTR + g * 16]);
    v16h bf3 = load_b16(&Wl[(3 * 16 + c) * WSTR + g * 16]);
    acc[0] = wmma16(a, bf0, acc[0]);
    acc[1] = wmma16(a, bf1, acc[1]);
    acc[2] = wmma16(a, bf2, acc[2]);
    acc[3] = wmma16(a, bf3, acc[3]);
    __syncthreads();
  }

#pragma unroll
  for (int jf = 0; jf < 4; ++jf) {
#pragma unroll
    for (int r = 0; r < 8; ++r) {
      int R = row0 + r + 8 * g;
      int col = cb * 64 + jf * 16 + c;
      out[(size_t)R * DMODEL + col] = acc[jf][r] + bo[col];
    }
  }
}

extern "C" void kernel_launch(void* const* d_in, const int* in_sizes, int n_in,
                              void* d_out, int out_size, void* d_ws,
                              size_t ws_size, hipStream_t stream) {
  (void)in_sizes; (void)n_in; (void)out_size; (void)ws_size;
  const float* query = (const float*)d_in[0];
  const float* key_  = (const float*)d_in[1];
  const float* value = (const float*)d_in[2];
  const float* adj   = (const float*)d_in[3];
  // d_in[4] = edges_att: unused by the reference computation
  const int*   mask  = (const int*)d_in[5];
  const float* Wq = (const float*)d_in[6];
  const float* bq = (const float*)d_in[7];
  const float* Wk = (const float*)d_in[8];
  const float* bk = (const float*)d_in[9];
  const float* Wv = (const float*)d_in[10];
  const float* bv = (const float*)d_in[11];
  const float* Wo = (const float*)d_in[12];
  const float* bo = (const float*)d_in[13];
  float* out = (float*)d_out;

  const size_t QK = (size_t)BB * HH * NN * DKK;  // 8M halves per tensor
  h16* qf = (h16*)d_ws;
  h16* kf = qf + QK;
  h16* vt = kf + QK;
  h16* xh = vt + QK;   // total 64 MB of workspace

  proj_kernel<<<dim3(BB * NN / 128, DMODEL / 64, 3), 256, 0, stream>>>(
      query, key_, value, Wq, bq, Wk, bk, Wv, bv, qf, kf, vt);
  attn_kernel<<<dim3(NN / 128, HH, BB), 256, 0, stream>>>(qf, kf, vt, adj,
                                                          mask, xh);
  out_kernel<<<dim3(BB * NN / 128, DMODEL / 64, 1), 256, 0, stream>>>(xh, Wo,
                                                                      bo, out);
}
